// TransformerLayer_84696755077831
// MI455X (gfx1250) — matrix operand order repaired and checked by reading, not compiled, not run on hardware
//
#include <hip/hip_runtime.h>
#include <cstdint>
#include <cstddef>

// ---------------------------------------------------------------------------
// Types
// ---------------------------------------------------------------------------
typedef __bf16 bf16;
typedef __attribute__((ext_vector_type(16))) __bf16 v16bf;
typedef __attribute__((ext_vector_type(8)))  __bf16 v8bf;
typedef __attribute__((ext_vector_type(4)))  __bf16 v4bf;
typedef __attribute__((ext_vector_type(8)))  float  v8f;

#define B_    2
#define S_    2048
#define D_    1024
#define H_    16
#define HD_   64
#define FF_   4096
#define MROWS (B_ * S_)   // 4096

__device__ __forceinline__ bf16 to_bf16(float f) {
    union { float f; uint32_t u; } x; x.f = f;
    uint32_t r = (x.u + 0x7FFFu + ((x.u >> 16) & 1u)) >> 16;
    union { uint16_t s; bf16 b; } y; y.s = (uint16_t)r;
    return y.b;
}

__device__ __forceinline__ v16bf cat16(v8bf lo, v8bf hi) {
    v16bf r;
#pragma unroll
    for (int i = 0; i < 8; ++i) { r[i] = lo[i]; r[i + 8] = hi[i]; }
    return r;
}

// ---------------------------------------------------------------------------
// CDNA5 async global->LDS copy (bypasses VGPRs, tracked by ASYNCcnt).
// Generic LDS pointers carry the wave-relative LDS byte offset in their low
// 32 bits (flat-aperture rule), so truncation yields the VDST operand.
// ---------------------------------------------------------------------------
__device__ __forceinline__ void async_copy_b128(void* lds_dst, const void* gsrc) {
    asm volatile("global_load_async_to_lds_b128 %0, %1, off"
                 :: "v"((uint32_t)(uintptr_t)lds_dst),
                    "v"((uint64_t)(uintptr_t)gsrc)
                 : "memory");
}
#define WAIT_ASYNC(n) asm volatile("s_wait_asynccnt %0" :: "n"(n) : "memory")

// ---------------------------------------------------------------------------
// f32 -> bf16 (vectorized, 4 elems/thread)
// ---------------------------------------------------------------------------
__global__ __launch_bounds__(256) void cvt_bf16_vec(const float* __restrict__ in,
                                                    bf16* __restrict__ out, int n4) {
    int i = blockIdx.x * 256 + threadIdx.x;
    if (i >= n4) return;
    const float4 f = ((const float4*)in)[i];
    v4bf o; o[0] = to_bf16(f.x); o[1] = to_bf16(f.y); o[2] = to_bf16(f.z); o[3] = to_bf16(f.w);
    *(v4bf*)(out + 4 * (size_t)i) = o;
}

// f32 [K][N] -> bf16 [N][K]  (LDS-tiled transpose)
__global__ __launch_bounds__(256) void cvt_bf16_T(const float* __restrict__ in,
                                                  bf16* __restrict__ out, int K, int N) {
    __shared__ float t[32][33];
    int n0 = blockIdx.x * 32, k0 = blockIdx.y * 32;
    int tx = threadIdx.x & 31, ty = threadIdx.x >> 5; // ty 0..7
#pragma unroll
    for (int i = 0; i < 32; i += 8)
        t[ty + i][tx] = in[(size_t)(k0 + ty + i) * N + n0 + tx];
    __syncthreads();
#pragma unroll
    for (int i = 0; i < 32; i += 8)
        out[(size_t)(n0 + ty + i) * K + k0 + tx] = to_bf16(t[tx][ty + i]);
}

// ---------------------------------------------------------------------------
// Tiled WMMA GEMM:  C[M,N] = A[M,K] * B[K,N] + bias,  B given as Bt[N][K]
// Block tile 128x128, BK=64 (2 wmma k-steps), 256 threads = 8 waves (2Mx4N),
// wave tile 64x32.  Double-buffered LDS filled by async global->LDS copies.
// mode 0: f32 out [M][N]
// mode 1: bf16 out, QKV head-split layout [B,H,S,HD]
// mode 2: bf16 out [M][N] with ReLU
// ---------------------------------------------------------------------------
#define BM 128
#define BN 128
#define BK 64

__global__ __launch_bounds__(256) void gemm_bf16_wmma(
    const bf16* __restrict__ A, const bf16* __restrict__ Bt,
    const float* __restrict__ bias, void* __restrict__ C,
    int M, int N, int K, int mode)
{
    __shared__ __align__(16) bf16 sA[2][BM][BK];
    __shared__ __align__(16) bf16 sB[2][BN][BK];

    const int tid  = threadIdx.x;
    const int lane = tid & 31;
    const int wave = tid >> 5;
    const int wm   = wave >> 2;          // 0..1
    const int wn   = wave & 3;           // 0..3
    const int half = lane >> 4;          // 0..1
    const int l15  = lane & 15;

    const int m0 = blockIdx.y * BM;
    const int n0 = blockIdx.x * BN;

    v8f acc[4][2];
#pragma unroll
    for (int i = 0; i < 4; ++i)
#pragma unroll
        for (int j = 0; j < 2; ++j)
#pragma unroll
            for (int e = 0; e < 8; ++e) acc[i][j][e] = 0.0f;

    // issue one tile's worth of async copies (8 x b128 per thread)
    auto issue_tile = [&](int k0, int buf) {
#pragma unroll
        for (int i = 0; i < 4; ++i) {
            int chunk = tid + i * 256;          // 0..1023
            int r = chunk >> 3, c = (chunk & 7) * 8;
            async_copy_b128(&sA[buf][r][c], &A [(size_t)(m0 + r) * K + k0 + c]);
            async_copy_b128(&sB[buf][r][c], &Bt[(size_t)(n0 + r) * K + k0 + c]);
        }
    };

    const int ntiles = K / BK;
    issue_tile(0, 0);

    for (int t = 0; t < ntiles; ++t) {
        const int buf = t & 1;
        if (t + 1 < ntiles) {
            issue_tile((t + 1) * BK, buf ^ 1);
            WAIT_ASYNC(8);        // 8 newest (tile t+1) may remain in flight
        } else {
            WAIT_ASYNC(0);
        }
        __syncthreads();

#pragma unroll
        for (int ks = 0; ks < 2; ++ks) {
            v16bf a[4], b[2];
#pragma unroll
            for (int i = 0; i < 4; ++i) {
                int row = wm * 64 + i * 16 + l15;
                v8bf lo = *(const v8bf*)&sA[buf][row][ks * 32 + half * 8];
                v8bf hi = *(const v8bf*)&sA[buf][row][ks * 32 + half * 8 + 16];
                a[i] = cat16(lo, hi);
            }
#pragma unroll
            for (int j = 0; j < 2; ++j) {
                int row = wn * 32 + j * 16 + l15;
                v8bf lo = *(const v8bf*)&sB[buf][row][ks * 32 + half * 8];
                v8bf hi = *(const v8bf*)&sB[buf][row][ks * 32 + half * 8 + 16];
                b[j] = cat16(lo, hi);
            }
#pragma unroll
            for (int i = 0; i < 4; ++i)
#pragma unroll
                for (int j = 0; j < 2; ++j)
                    acc[i][j] = __builtin_amdgcn_wmma_f32_16x16x32_bf16(
                        false, a[i], false, b[j], (short)0, acc[i][j], false, false);
        }
        __syncthreads();   // buffer (t&1) free for reuse by iteration t+1's issue
    }

    // epilogue
    float bj[2];
#pragma unroll
    for (int j = 0; j < 2; ++j) bj[j] = bias[n0 + wn * 32 + j * 16 + l15];

#pragma unroll
    for (int i = 0; i < 4; ++i) {
#pragma unroll
        for (int j = 0; j < 2; ++j) {
            int n = n0 + wn * 32 + j * 16 + l15;
#pragma unroll
            for (int r = 0; r < 8; ++r) {
                int m = m0 + wm * 64 + i * 16 + r + 8 * half;
                float v = acc[i][j][r] + bj[j];
                if (mode == 0) {
                    ((float*)C)[(size_t)m * N + n] = v;
                } else if (mode == 1) {
                    int bb = m >> 11, s = m & (S_ - 1);
                    int h = n >> 6,  hd = n & (HD_ - 1);
                    ((bf16*)C)[((((size_t)(bb * H_ + h)) * S_ + s) << 6) + hd] = to_bf16(v);
                } else {
                    ((bf16*)C)[(size_t)m * N + n] = to_bf16(v > 0.0f ? v : 0.0f);
                }
            }
        }
    }
}

// ---------------------------------------------------------------------------
// Flash attention (causal).  Q,K,V bf16 [B,H,S,HD]; ctx bf16 [B,S,D].
// grid: (S/64, B*H), block 128 (4 waves, 16 q-rows each).
// K tile staged via async global->LDS; V transposed through VGPRs.
// ---------------------------------------------------------------------------
__global__ __launch_bounds__(128) void flash_attn_wmma(
    const bf16* __restrict__ Q, const bf16* __restrict__ Kb,
    const bf16* __restrict__ Vb, bf16* __restrict__ Ctx)
{
    __shared__ __align__(16) bf16 sK[32][64];   // [key][hd]
    __shared__ __align__(16) bf16 sV[64][32];   // [hd][key] (transposed)
    __shared__ __align__(16) bf16 sP[4][16][32];// per-wave P scratch [m][key]

    const int qblk = blockIdx.x;
    const int bh   = blockIdx.y;
    const int bIdx = bh >> 4, h = bh & 15;
    const int q0   = qblk * 64;

    const int tid  = threadIdx.x;
    const int lane = tid & 31;
    const int wave = tid >> 5;
    const int half = lane >> 4;
    const int l15  = lane & 15;

    const size_t baseBH = ((size_t)(bIdx * H_ + h)) * S_ * HD_;

    // Q fragments (16 rows x 64, two k-steps of 32)
    v16bf qf[2];
    {
        int qrow = q0 + wave * 16 + l15;
        const bf16* qp = Q + baseBH + (size_t)qrow * HD_;
#pragma unroll
        for (int ks = 0; ks < 2; ++ks) {
            v8bf lo = *(const v8bf*)(qp + ks * 32 + half * 8);
            v8bf hi = *(const v8bf*)(qp + ks * 32 + half * 8 + 16);
            qf[ks] = cat16(lo, hi);
        }
    }

    float mrow[8], lrow[8];
    v8f o[4];
#pragma unroll
    for (int r = 0; r < 8; ++r) { mrow[r] = -1e30f; lrow[r] = 0.0f; }
#pragma unroll
    for (int ot = 0; ot < 4; ++ot)
#pragma unroll
        for (int e = 0; e < 8; ++e) o[ot][e] = 0.0f;

    const int kmax = q0 + 64;
    for (int kb = 0; kb < kmax; kb += 32) {
        // K tile: async copy straight into LDS. V tile: transpose via VGPRs.
#pragma unroll
        for (int i = 0; i < 2; ++i) {
            int chunk = tid + i * 128;           // 0..255
            int r = chunk >> 3, c = (chunk & 7) * 8;
            async_copy_b128(&sK[r][c], &Kb[baseBH + (size_t)(kb + r) * HD_ + c]);
            v8bf vv = *(const v8bf*)&Vb[baseBH + (size_t)(kb + r) * HD_ + c];
#pragma unroll
            for (int e = 0; e < 8; ++e) sV[c + e][r] = vv[e];
        }
        WAIT_ASYNC(0);
        __syncthreads();

        // scores: 16 q-rows x 32 keys (2 N-tiles, 2 k-steps over HD)
        v8f sc[2];
#pragma unroll
        for (int nt = 0; nt < 2; ++nt) {
#pragma unroll
            for (int e = 0; e < 8; ++e) sc[nt][e] = 0.0f;
            int row = nt * 16 + l15;
#pragma unroll
            for (int ks = 0; ks < 2; ++ks) {
                v8bf lo = *(const v8bf*)&sK[row][ks * 32 + half * 8];
                v8bf hi = *(const v8bf*)&sK[row][ks * 32 + half * 8 + 16];
                sc[nt] = __builtin_amdgcn_wmma_f32_16x16x32_bf16(
                    false, qf[ks], false, cat16(lo, hi), (short)0, sc[nt], false, false);
            }
        }

        // online softmax update (finite sentinel avoids inf-inf NaN)
        float mnew[8], oscale[8];
#pragma unroll
        for (int r = 0; r < 8; ++r) {
            int qrow = q0 + wave * 16 + r + 8 * half;
            float mx = mrow[r];
#pragma unroll
            for (int nt = 0; nt < 2; ++nt) {
                int key = kb + nt * 16 + l15;
                float s = sc[nt][r] * 0.125f;           // 1/sqrt(64)
                s = (key <= qrow) ? s : -1e30f;
                sc[nt][r] = s;
                mx = fmaxf(mx, s);
            }
#pragma unroll
            for (int msk = 1; msk < 16; msk <<= 1)
                mx = fmaxf(mx, __shfl_xor(mx, msk, 32));
            mnew[r] = mx;
        }
#pragma unroll
        for (int r = 0; r < 8; ++r) {
            float mn = mnew[r];
            float rs = 0.0f;
#pragma unroll
            for (int nt = 0; nt < 2; ++nt) {
                float p = __expf(sc[nt][r] - mn);
                sc[nt][r] = p;
                rs += p;
            }
#pragma unroll
            for (int msk = 1; msk < 16; msk <<= 1)
                rs += __shfl_xor(rs, msk, 32);
            float a_ = __expf(mrow[r] - mn);
            lrow[r] = lrow[r] * a_ + rs;
            mrow[r] = mn;
            oscale[r] = a_;
        }
#pragma unroll
        for (int ot = 0; ot < 4; ++ot)
#pragma unroll
            for (int r = 0; r < 8; ++r) o[ot][r] *= oscale[r];

        // reshape P (C-layout) -> A-layout via LDS
#pragma unroll
        for (int nt = 0; nt < 2; ++nt)
#pragma unroll
            for (int r = 0; r < 8; ++r)
                sP[wave][r + 8 * half][nt * 16 + l15] = to_bf16(sc[nt][r]);
        __syncthreads();

        v16bf pf;
        {
            v8bf lo = *(const v8bf*)&sP[wave][l15][half * 8];
            v8bf hi = *(const v8bf*)&sP[wave][l15][16 + half * 8];
            pf = cat16(lo, hi);
        }
#pragma unroll
        for (int ot = 0; ot < 4; ++ot) {
            int row = ot * 16 + l15;
            v8bf lo = *(const v8bf*)&sV[row][half * 8];
            v8bf hi = *(const v8bf*)&sV[row][half * 8 + 16];
            o[ot] = __builtin_amdgcn_wmma_f32_16x16x32_bf16(
                false, pf, false, cat16(lo, hi), (short)0, o[ot], false, false);
        }
        __syncthreads();   // protect sK/sV/sP before next tile load
    }

    // normalize and write ctx as [B,S,D] (head-merged)
#pragma unroll
    for (int ot = 0; ot < 4; ++ot) {
#pragma unroll
        for (int r = 0; r < 8; ++r) {
            int srow = q0 + wave * 16 + r + 8 * half;
            size_t off = ((size_t)bIdx * S_ + srow) * D_ + h * HD_ + ot * 16 + l15;
            Ctx[off] = to_bf16(o[ot][r] / lrow[r]);
        }
    }
}

// ---------------------------------------------------------------------------
// Fused residual + LayerNorm: out = LN(X + Y) * g + b ; optional bf16 copy
// grid: MROWS rows, block 256, 4 elems/thread (D=1024)
// ---------------------------------------------------------------------------
__global__ __launch_bounds__(256) void add_ln_kernel(
    const float* __restrict__ X, const float* __restrict__ Y,
    const float* __restrict__ g, const float* __restrict__ be,
    float* __restrict__ outF, bf16* __restrict__ outB)
{
    const int row = blockIdx.x;
    const int tid = threadIdx.x;
    const float* x = X + (size_t)row * D_;
    const float* y = Y + (size_t)row * D_;

    float v[4], s = 0.0f, sq = 0.0f;
#pragma unroll
    for (int i = 0; i < 4; ++i) {
        int c = tid + i * 256;
        v[i] = x[c] + y[c];
        s += v[i];
        sq += v[i] * v[i];
    }
#pragma unroll
    for (int m = 1; m < 32; m <<= 1) {
        s  += __shfl_xor(s,  m, 32);
        sq += __shfl_xor(sq, m, 32);
    }
    __shared__ float ss[8], ssq[8];
    if ((tid & 31) == 0) { ss[tid >> 5] = s; ssq[tid >> 5] = sq; }
    __syncthreads();
    s = 0.0f; sq = 0.0f;
#pragma unroll
    for (int w = 0; w < 8; ++w) { s += ss[w]; sq += ssq[w]; }

    const float mean = s * (1.0f / D_);
    const float var  = sq * (1.0f / D_) - mean * mean;
    const float rstd = rsqrtf(var + 1e-5f);

#pragma unroll
    for (int i = 0; i < 4; ++i) {
        int c = tid + i * 256;
        float o = (v[i] - mean) * rstd * g[c] + be[c];
        outF[(size_t)row * D_ + c] = o;
        if (outB) outB[(size_t)row * D_ + c] = to_bf16(o);
    }
}

// ---------------------------------------------------------------------------
// Host orchestration
// ---------------------------------------------------------------------------
extern "C" void kernel_launch(void* const* d_in, const int* in_sizes, int n_in,
                              void* d_out, int out_size, void* d_ws, size_t ws_size,
                              hipStream_t stream) {
    (void)in_sizes; (void)n_in; (void)out_size; (void)ws_size;

    const float* src = (const float*)d_in[0];
    // d_in[1] = causal_mask (unused; causality applied analytically)
    const float* Wq = (const float*)d_in[2];  const float* bq = (const float*)d_in[3];
    const float* Wk = (const float*)d_in[4];  const float* bk = (const float*)d_in[5];
    const float* Wv = (const float*)d_in[6];  const float* bv = (const float*)d_in[7];
    const float* Wo = (const float*)d_in[8];  const float* bo = (const float*)d_in[9];
    const float* W1 = (const float*)d_in[10]; const float* b1 = (const float*)d_in[11];
    const float* W2 = (const float*)d_in[12]; const float* b2 = (const float*)d_in[13];
    const float* g1 = (const float*)d_in[14]; const float* be1 = (const float*)d_in[15];
    const float* g2 = (const float*)d_in[16]; const float* be2 = (const float*)d_in[17];

    // workspace carve-up (aliased; ~96 MB)
    char* p = (char*)d_ws;
    auto carve = [&](size_t bytes) -> char* {
        char* r = p;
        p += (bytes + 255) & ~(size_t)255;
        return r;
    };
    const size_t MD2 = (size_t)MROWS * D_ * 2;      // 8 MB (bf16 [4096,1024])
    const size_t MD4 = (size_t)MROWS * D_ * 4;      // 16 MB (f32  [4096,1024])

    bf16* sSrc = (bf16*)carve(MD2);                 // later reused as sX1
    bf16* sWq  = (bf16*)carve((size_t)D_ * D_ * 2);
    bf16* sWk  = (bf16*)carve((size_t)D_ * D_ * 2);
    bf16* sWv  = (bf16*)carve((size_t)D_ * D_ * 2);
    bf16* sWo  = (bf16*)carve((size_t)D_ * D_ * 2);
    bf16* sW1t = (bf16*)carve((size_t)D_ * FF_ * 2);
    bf16* sW2t = (bf16*)carve((size_t)FF_ * D_ * 2);
    bf16* sQ   = (bf16*)carve(MD2);                 // sQ..sCtx region reused as sFF1
    bf16* sK   = (bf16*)carve(MD2);
    bf16* sV   = (bf16*)carve(MD2);
    bf16* sCtx = (bf16*)carve(MD2);
    float* fAttn = (float*)carve(MD4);              // later reused as fFF2
    float* fX1   = (float*)carve(MD4);

    bf16*  sX1  = sSrc;    // src bf16 dead after QKV projections
    bf16*  sFF1 = sQ;      // 32 MB contiguous: sQ+sK+sV+sCtx, dead after O-proj
    float* fFF2 = fAttn;   // dead after ln1

    // 1) precision conversion
    cvt_bf16_vec<<<(MROWS * D_) / 1024, 256, 0, stream>>>(src, sSrc, (MROWS * D_) / 4);
    cvt_bf16_T<<<dim3(D_ / 32, D_ / 32),  256, 0, stream>>>(Wq, sWq, D_, D_);
    cvt_bf16_T<<<dim3(D_ / 32, D_ / 32),  256, 0, stream>>>(Wk, sWk, D_, D_);
    cvt_bf16_T<<<dim3(D_ / 32, D_ / 32),  256, 0, stream>>>(Wv, sWv, D_, D_);
    cvt_bf16_T<<<dim3(D_ / 32, D_ / 32),  256, 0, stream>>>(Wo, sWo, D_, D_);
    cvt_bf16_T<<<dim3(FF_ / 32, D_ / 32), 256, 0, stream>>>(W1, sW1t, D_, FF_);
    cvt_bf16_T<<<dim3(D_ / 32, FF_ / 32), 256, 0, stream>>>(W2, sW2t, FF_, D_);

    const dim3 gD(D_ / BN, MROWS / BM);    // (8, 32)
    const dim3 gFF(FF_ / BN, MROWS / BM);  // (32, 32)

    // 2) QKV projections -> [B,H,S,HD] bf16
    gemm_bf16_wmma<<<gD, 256, 0, stream>>>(sSrc, sWq, bq, sQ, MROWS, D_, D_, 1);
    gemm_bf16_wmma<<<gD, 256, 0, stream>>>(sSrc, sWk, bk, sK, MROWS, D_, D_, 1);
    gemm_bf16_wmma<<<gD, 256, 0, stream>>>(sSrc, sWv, bv, sV, MROWS, D_, D_, 1);

    // 3) causal flash attention -> ctx bf16 [B,S,D]
    flash_attn_wmma<<<dim3(S_ / 64, B_ * H_), 128, 0, stream>>>(sQ, sK, sV, sCtx);

    // 4) output projection -> f32
    gemm_bf16_wmma<<<gD, 256, 0, stream>>>(sCtx, sWo, bo, fAttn, MROWS, D_, D_, 0);

    // 5) x1 = LN(src + attn_out)   (f32 + bf16 copies)
    add_ln_kernel<<<MROWS, 256, 0, stream>>>(src, fAttn, g1, be1, fX1, sX1);

    // 6) FFN
    gemm_bf16_wmma<<<gFF, 256, 0, stream>>>(sX1, sW1t, b1, sFF1, MROWS, FF_, D_, 2);
    gemm_bf16_wmma<<<gD,  256, 0, stream>>>(sFF1, sW2t, b2, fFF2, MROWS, D_, FF_, 0);

    // 7) out = LN(x1 + ff)
    add_ln_kernel<<<MROWS, 256, 0, stream>>>(fX1, fFF2, g2, be2, (float*)d_out, nullptr);
}
